// self_Attention_50766513439213
// MI455X (gfx1250) — matrix operand order repaired and checked by reading, not compiled, not run on hardware
//
#include <hip/hip_runtime.h>
#include <hip/hip_bf16.h>

typedef __attribute__((ext_vector_type(16))) __bf16        v16bf;
typedef __attribute__((ext_vector_type(8)))  float         v8f;
typedef __attribute__((ext_vector_type(4)))  unsigned int  v4u;

#define WMMA_BF16(a, b, c) \
  __builtin_amdgcn_wmma_f32_16x16x32_bf16(false, (a), false, (b), (short)0, (c), false, false)

// Native f32->bf16 (hardware cvt, RTNE)
__device__ __forceinline__ unsigned short f2bf(float f) {
  union { __bf16 h; unsigned short u; } r; r.h = (__bf16)f; return r.u;
}
__device__ __forceinline__ unsigned int pack2bf(float a, float b) {
  union { __bf16 h[2]; unsigned int u; } r;
  r.h[0] = (__bf16)a; r.h[1] = (__bf16)b;
  return r.u;
}
__device__ __forceinline__ float bf2f(unsigned short s) {
  return __uint_as_float(((unsigned int)s) << 16);
}

union FragBF { v16bf v; unsigned short u[16]; v4u q[2]; };

__device__ __forceinline__ v8f vzero() {
  v8f z;
#pragma unroll
  for (int i = 0; i < 8; ++i) z[i] = 0.0f;
  return z;
}

// CDNA5 async global->LDS copy of one 16B chunk per lane (ASYNCcnt-tracked).
// LDS destination = wave-relative byte offset (low 32 bits of generic pointer).
__device__ __forceinline__ void async_cp16(unsigned short* lds_dst, const unsigned short* g_src) {
  unsigned int off = (unsigned int)(size_t)lds_dst;
  asm volatile("global_load_async_to_lds_b128 %0, %1, off"
               :: "v"(off), "v"(g_src) : "memory");
}
__device__ __forceinline__ void async_wait0() {
  asm volatile("s_wait_asynccnt 0x0" ::: "memory");
}

// A fragment: 16x32 bf16 (rows=M, cols=K) from LDS row-major ushort.
// lane m = lane&15, half = lane>>4; u[0..15]=K[16h..16h+15]
__device__ __forceinline__ v16bf load_a_frag(const unsigned short* s, int row0, int stride, int col0) {
  int lane = threadIdx.x & 31;
  const unsigned short* p = s + (size_t)(row0 + (lane & 15)) * stride + col0 + 16 * (lane >> 4);
  FragBF r;
#pragma unroll
  for (int i = 0; i < 8; ++i) r.u[i] = p[i];
#pragma unroll
  for (int i = 0; i < 8; ++i) r.u[8 + i] = p[8 + i];
  return r.v;
}

// B fragment: 32x16 bf16 (rows=K, cols=N) from LDS row-major ushort.
__device__ __forceinline__ v16bf load_b_frag_lds(const unsigned short* s, int k0, int stride, int col0) {
  int lane = threadIdx.x & 31;
  const unsigned short* p = s + (size_t)(k0 + 16 * (lane >> 4)) * stride + col0 + (lane & 15);
  FragBF r;
#pragma unroll
  for (int i = 0; i < 16; ++i) r.u[i] = p[(size_t)i * stride];
  return r.v;
}

// B fragment where logical B is the TRANSPOSE of an LDS tile: B[k][n] = s[n][k].
__device__ __forceinline__ v16bf load_bT_frag_lds(const unsigned short* s, int k0, int stride, int col0) {
  int lane = threadIdx.x & 31;
  const unsigned short* p = s + (size_t)(col0 + (lane & 15)) * stride + k0 + 16 * (lane >> 4);
  FragBF r;
#pragma unroll
  for (int i = 0; i < 16; ++i) r.u[i] = p[i];
  return r.v;
}

// B fragment from pre-swizzled bf16 weights: per lane 32B contiguous -> 2x global_load_b128.
__device__ __forceinline__ v16bf load_b_frag_sw(const unsigned short* Bsw, int ktiles, int ntile, int kt) {
  int lane = threadIdx.x & 31;
  const v4u* p = (const v4u*)(Bsw + (((size_t)ntile * ktiles + kt) * 32 + lane) * 16);
  FragBF r;
  r.q[0] = p[0];
  r.q[1] = p[1];
  return r.v;
}

// ---------------------------------------------------------------------------
// One-shot: row-major f32 weights [K,N] -> swizzled bf16 fragment order.
// ---------------------------------------------------------------------------
__global__ __launch_bounds__(256) void swizzle_weights(const float* __restrict__ src,
                                                       unsigned short* __restrict__ dst,
                                                       int K, int N) {
  const int g = blockIdx.x * 256 + threadIdx.x;
  const int lane = g & 31;
  const int t = g >> 5;
  const int ktiles = K >> 5;
  const int kt = t % ktiles, nt = t / ktiles;
  const int n = lane & 15, half = lane >> 4;
  const float* s = src + (size_t)(kt * 32 + 16 * half) * N + nt * 16 + n;
  unsigned short* d = dst + (size_t)g * 16;
#pragma unroll
  for (int i = 0; i < 16; ++i) d[i] = f2bf(s[(size_t)i * N]);
}

// ---------------------------------------------------------------------------
// GEMM (A f32 -> staged bf16, B pre-swizzled bf16, C stored bf16).
// Block tile 128x64, 8 waves, K-chunk 64. M%128==0, N%64==0, K%64==0.
// ---------------------------------------------------------------------------
__global__ __launch_bounds__(256) void gemm_af32_cbf16(const float* __restrict__ A,
                                                       const unsigned short* __restrict__ Bsw,
                                                       unsigned short* __restrict__ C16,
                                                       int K, int lda, int ldc) {
  __shared__ unsigned short As[128 * 72];
  const int tid = threadIdx.x;
  const int bm = blockIdx.y * 128, bn = blockIdx.x * 64;
  const int w = tid >> 5, lane = tid & 31;
  const int nt = w & 3, mg = w >> 2;
  const int ntile = (bn >> 4) + nt;
  const int ktiles = K >> 5;

  v8f acc[4];
#pragma unroll
  for (int i = 0; i < 4; ++i) acc[i] = vzero();

  for (int kc = 0; kc < K; kc += 64) {
    for (int e = tid; e < 128 * 16; e += 256) {
      int r = e >> 4, c4 = (e & 15) << 2;
      const float4 f = *(const float4*)&A[(size_t)(bm + r) * lda + kc + c4];
      *(unsigned int*)&As[r * 72 + c4]     = pack2bf(f.x, f.y);
      *(unsigned int*)&As[r * 72 + c4 + 2] = pack2bf(f.z, f.w);
    }
    if (kc + 64 < K) {   // prefetch next A chunk (global_prefetch_b8)
      int r = tid >> 1, seg = tid & 1;
      __builtin_prefetch(&A[(size_t)(bm + r) * lda + kc + 64 + seg * 32], 0, 3);
    }
    __syncthreads();
#pragma unroll
    for (int ks = 0; ks < 64; ks += 32) {
      v16bf bf = load_b_frag_sw(Bsw, ktiles, ntile, (kc + ks) >> 5);
#pragma unroll
      for (int i = 0; i < 4; ++i) {
        v16bf af = load_a_frag(As, (mg * 4 + i) * 16, 72, ks);
        acc[i] = WMMA_BF16(af, bf, acc[i]);
      }
    }
    __syncthreads();
  }
  const int hlf = lane >> 4, n = lane & 15;
  const int coln = bn + nt * 16 + n;
#pragma unroll
  for (int i = 0; i < 4; ++i) {
    int mrow = bm + (mg * 4 + i) * 16;
#pragma unroll
    for (int j = 0; j < 8; ++j)
      C16[(size_t)(mrow + j + 8 * hlf) * ldc + coln] = f2bf(acc[i][j]);
  }
}

// ---------------------------------------------------------------------------
// GEMM (A already bf16: staged via async global->LDS copies; C stored f32).
// ---------------------------------------------------------------------------
__global__ __launch_bounds__(256) void gemm_abf16_cf32(const unsigned short* __restrict__ A16,
                                                       const unsigned short* __restrict__ Bsw,
                                                       float* __restrict__ C,
                                                       int K, int lda, int ldc) {
  __shared__ unsigned short As[128 * 72];
  const int tid = threadIdx.x;
  const int bm = blockIdx.y * 128, bn = blockIdx.x * 64;
  const int w = tid >> 5, lane = tid & 31;
  const int nt = w & 3, mg = w >> 2;
  const int ntile = (bn >> 4) + nt;
  const int ktiles = K >> 5;

  v8f acc[4];
#pragma unroll
  for (int i = 0; i < 4; ++i) acc[i] = vzero();

  for (int kc = 0; kc < K; kc += 64) {
    // async copy A chunk (128 rows x 64 bf16): 1024 16B chunks
    for (int e = tid; e < 128 * 8; e += 256) {
      int r = e >> 3, c8 = (e & 7) << 3;
      async_cp16(&As[r * 72 + c8], &A16[(size_t)(bm + r) * lda + kc + c8]);
    }
    if (kc + 64 < K && tid < 128)   // next chunk row = 128B = one cacheline
      __builtin_prefetch(&A16[(size_t)(bm + tid) * lda + kc + 64], 0, 3);
    async_wait0();
    __syncthreads();
#pragma unroll
    for (int ks = 0; ks < 64; ks += 32) {
      v16bf bf = load_b_frag_sw(Bsw, ktiles, ntile, (kc + ks) >> 5);
#pragma unroll
      for (int i = 0; i < 4; ++i) {
        v16bf af = load_a_frag(As, (mg * 4 + i) * 16, 72, ks);
        acc[i] = WMMA_BF16(af, bf, acc[i]);
      }
    }
    __syncthreads();
  }
  const int hlf = lane >> 4, n = lane & 15;
  const int coln = bn + nt * 16 + n;
#pragma unroll
  for (int i = 0; i < 4; ++i) {
    int mrow = bm + (mg * 4 + i) * 16;
#pragma unroll
    for (int j = 0; j < 8; ++j)
      C[(size_t)(mrow + j + 8 * hlf) * ldc + coln] = acc[i][j];
  }
}

// ---------------------------------------------------------------------------
// attn logits: a_pre[b,n,h*64+m] = bf16( 0.125 * sum_d q[n,hd+d]*k[m,hd+d] )
// q16/kv16 are bf16; tiles staged with async global->LDS copies.
// ---------------------------------------------------------------------------
__global__ __launch_bounds__(256) void attn_logits(const unsigned short* __restrict__ q16,
                                                   const unsigned short* __restrict__ kv16,
                                                   unsigned short* __restrict__ apre16) {
  __shared__ unsigned short qh[32 * 72];
  __shared__ unsigned short kh[64 * 72];
  const int b = blockIdx.x, tid = threadIdx.x;
  const int w = tid >> 5, lane = tid & 31;
  const int mt = w >> 2, nt = w & 3;
  const unsigned short* qb = q16 + (size_t)b * 17 * 512;
  const unsigned short* kb = kv16 + (size_t)b * 64 * 1024;   // k = cols [0,512)

  for (int h = 0; h < 8; ++h) {
    __syncthreads();
    // qh: rows 0..16 async copy, rows 17..31 zero (256 chunks of 16B)
    {
      int e = tid;                       // exactly 32*8 chunks
      int r = e >> 3, c8 = (e & 7) << 3;
      if (r < 17) {
        async_cp16(&qh[r * 72 + c8], &qb[r * 512 + h * 64 + c8]);
      } else {
        unsigned long long* z = (unsigned long long*)&qh[r * 72 + c8];
        z[0] = 0ull; z[1] = 0ull;
      }
    }
    // kh: 64x64 bf16 = 512 chunks
    for (int e = tid; e < 64 * 8; e += 256) {
      int r = e >> 3, c8 = (e & 7) << 3;
      async_cp16(&kh[r * 72 + c8], &kb[r * 1024 + h * 64 + c8]);
    }
    async_wait0();
    __syncthreads();
    v8f acc = vzero();
#pragma unroll
    for (int ks = 0; ks < 64; ks += 32) {
      v16bf af = load_a_frag(qh, mt * 16, 72, ks);
      v16bf bf = load_bT_frag_lds(kh, ks, 72, nt * 16);   // B = k^T
      acc = WMMA_BF16(af, bf, acc);
    }
    const int hlf = lane >> 4, mcol = nt * 16 + (lane & 15);
#pragma unroll
    for (int j = 0; j < 8; ++j) {
      int row = mt * 16 + j + 8 * hlf;
      if (row < 17)
        apre16[(size_t)(b * 17 + row) * 512 + h * 64 + mcol] = f2bf(acc[j] * 0.125f);
    }
  }
}

// ---------------------------------------------------------------------------
// BatchNorm stats over 17408 rows -> per-channel scale/shift (biased var)
// ---------------------------------------------------------------------------
__global__ __launch_bounds__(256) void bn_stats(const float* __restrict__ a,
                                                const float* __restrict__ gamma,
                                                const float* __restrict__ beta,
                                                float* __restrict__ scale,
                                                float* __restrict__ shift) {
  __shared__ float ls[4][64], ls2[4][64];
  const int cx = threadIdx.x & 63, ry = threadIdx.x >> 6;
  const int c = blockIdx.x * 64 + cx;
  float s = 0.f, s2 = 0.f;
  for (int r = ry; r < 17408; r += 4) {
    float v = a[(size_t)r * 512 + c];
    s += v; s2 += v * v;
  }
  ls[ry][cx] = s; ls2[ry][cx] = s2;
  __syncthreads();
  if (ry == 0) {
    float sum  = ls[0][cx] + ls[1][cx] + ls[2][cx] + ls[3][cx];
    float sum2 = ls2[0][cx] + ls2[1][cx] + ls2[2][cx] + ls2[3][cx];
    const float invn = 1.0f / 17408.0f;
    float mean = sum * invn;
    float var  = sum2 * invn - mean * mean;
    float inv  = rsqrtf(var + 1e-5f);
    float sc   = gamma[c] * inv;
    scale[c] = sc;
    shift[c] = beta[c] - mean * sc;
  }
}

// COCO skeleton adjacency (incl. self), bit m set => joint m is a neighbor of n
static __device__ const unsigned int ADJ[17] = {
  0x7u, 0xFu, 0x17u, 0x2Au, 0x54u, 0x8E8u, 0x1170u, 0x2A0u, 0x540u,
  0x280u, 0x500u, 0x3820u, 0x5840u, 0xA800u, 0x15000u, 0xA000u, 0x14000u};

// ---------------------------------------------------------------------------
// Fused tail: BN apply -> adjacency -> softmax -> x = attn@v -> out = x@Wp+bp
// ---------------------------------------------------------------------------
__global__ __launch_bounds__(256) void fused_out(const float* __restrict__ a,
                                                 const unsigned short* __restrict__ kv16,
                                                 const float* __restrict__ scale,
                                                 const float* __restrict__ shift,
                                                 const unsigned short* __restrict__ WpSw,
                                                 const float* __restrict__ bp,
                                                 float* __restrict__ out) {
  __shared__ unsigned short smA[32 * 520];  // attn (softmaxed bf16); later x_bf
  __shared__ unsigned short smB[17 * 520];  // a_bn (bf16); later v_h tile (64x72)
  const int b = blockIdx.x, tid = threadIdx.x;
  const int w = tid >> 5, lane = tid & 31;
  const int hlf = lane >> 4, ln = lane & 15;
  const float* ab = a + (size_t)b * 17 * 512;

  // 1: BatchNorm apply -> bf16 LDS (paired)
  for (int e = tid; e < 17 * 256; e += 256) {
    int r = e >> 8, c = (e & 255) << 1;
    const float2 av = *(const float2*)&ab[r * 512 + c];
    const float2 sc = *(const float2*)&scale[c];
    const float2 sh = *(const float2*)&shift[c];
    *(unsigned int*)&smB[r * 520 + c] = pack2bf(av.x * sc.x + sh.x, av.y * sc.y + sh.y);
  }
  // zero padding rows 17..31 of attn buffer (WMMA A operand)
  for (int e = tid; e < 15 * 256; e += 256) {
    int r = 17 + (e >> 8), c = (e & 255) << 1;
    *(unsigned int*)&smA[r * 520 + c] = 0u;
  }
  __syncthreads();

  // 2: adjacency aggregation + softmax over m for each (joint n, head h)
  if (tid < 136) {
    const int n = tid >> 3, h = tid & 7;
    const unsigned int mask = ADJ[n];
    const int cbase = h * 64;
    float mx = -3.4e38f;
    for (int d = 0; d < 64; ++d) {
      float acc = 0.f;
      for (int m = 0; m < 17; ++m)
        if ((mask >> m) & 1u) acc += bf2f(smB[m * 520 + cbase + d]);
      mx = fmaxf(mx, acc);
    }
    float ssum = 0.f;
    for (int d = 0; d < 64; ++d) {
      float acc = 0.f;
      for (int m = 0; m < 17; ++m)
        if ((mask >> m) & 1u) acc += bf2f(smB[m * 520 + cbase + d]);
      float e = __expf(acc - mx);
      ssum += e;
      smA[n * 520 + cbase + d] = f2bf(e);
    }
    float rinv = 1.0f / ssum;
    for (int d = 0; d < 64; ++d)
      smA[n * 520 + cbase + d] = f2bf(bf2f(smA[n * 520 + cbase + d]) * rinv);
  }

  // 3: per-head x = attn @ v  (async-staged bf16 v tiles; x acc in VGPRs)
  const unsigned short* kvb = kv16 + (size_t)b * 64 * 1024;   // v = cols [512,1024)
  const int mt = w >> 2, dt = w & 3;
  v8f xacc[8];
#pragma unroll
  for (int h = 0; h < 8; ++h) xacc[h] = vzero();

  for (int h = 0; h < 8; ++h) {
    __syncthreads();   // protects smB reuse + (h==0) attn writes
    for (int e = tid; e < 64 * 8; e += 256) {     // 512 chunks of 16B
      int r = e >> 3, c8 = (e & 7) << 3;
      async_cp16(&smB[r * 72 + c8], &kvb[r * 1024 + 512 + h * 64 + c8]);
    }
    async_wait0();
    __syncthreads();
#pragma unroll
    for (int ks = 0; ks < 64; ks += 32) {
      v16bf af = load_a_frag(smA, mt * 16, 520, h * 64 + ks);
      v16bf bf = load_b_frag_lds(smB, ks, 72, dt * 16);
      xacc[h] = WMMA_BF16(af, bf, xacc[h]);
    }
  }
  __syncthreads();

  // 4: write x (bf16) into smA (rows 17..31 stay exact zeros)
#pragma unroll
  for (int h = 0; h < 8; ++h) {
#pragma unroll
    for (int j = 0; j < 8; ++j) {
      int row = mt * 16 + j + 8 * hlf;
      smA[row * 520 + h * 64 + dt * 16 + ln] = f2bf(xacc[h][j]);
    }
  }
  __syncthreads();

  // 5: out = x @ Wp + bp ; each wave owns 4 N-subtiles x 2 M-subtiles
  for (int i = 0; i < 4; ++i) {
    const int ntile = w * 4 + i;
    const int col0 = ntile * 16;
    v8f oacc[2] = {vzero(), vzero()};
    for (int kc = 0; kc < 512; kc += 32) {
      v16bf bf = load_b_frag_sw(WpSw, 16, ntile, kc >> 5);
#pragma unroll
      for (int m = 0; m < 2; ++m) {
        v16bf af = load_a_frag(smA, m * 16, 520, kc);
        oacc[m] = WMMA_BF16(af, bf, oacc[m]);
      }
    }
#pragma unroll
    for (int m = 0; m < 2; ++m) {
#pragma unroll
      for (int j = 0; j < 8; ++j) {
        int row = m * 16 + j + 8 * hlf;
        if (row < 17)
          out[(size_t)(b * 17 + row) * 512 + col0 + ln] = oacc[m][j] + bp[col0 + ln];
      }
    }
  }
}

// ---------------------------------------------------------------------------
extern "C" void kernel_launch(void* const* d_in, const int* in_sizes, int n_in,
                              void* d_out, int out_size, void* d_ws, size_t ws_size,
                              hipStream_t stream) {
  const float* noise = (const float*)d_in[0];  // [1024,17,512]
  const float* obj   = (const float*)d_in[1];  // [1024,64,512]
  const float* W     = (const float*)d_in[2];  // [512,512]
  const float* gamma = (const float*)d_in[3];  // [512]
  const float* beta  = (const float*)d_in[4];  // [512]
  const float* Wkv   = (const float*)d_in[5];  // [512,1024]
  const float* Wq    = (const float*)d_in[6];  // [512,512]
  const float* Wp    = (const float*)d_in[7];  // [512,512]
  const float* bp    = (const float*)d_in[8];  // [512]
  float* out = (float*)d_out;

  const size_t NQ  = (size_t)17408 * 512;   // B*17 x 512
  const size_t NKV = (size_t)65536 * 1024;  // B*64 x 1024
  const size_t SW  = (size_t)512 * 512 * 3 + (size_t)512 * 1024;  // swizzled weights
  const size_t need = (NQ + 1024) * sizeof(float) +
                      (2 * NQ + NKV + SW) * sizeof(unsigned short);
  if (ws_size < need) return;

  float* ws_a     = (float*)d_ws;                       // f32 a (for BN stats)
  float* ws_scale = ws_a + NQ;
  float* ws_shift = ws_scale + 512;
  unsigned short* q16    = (unsigned short*)(ws_shift + 512);
  unsigned short* kv16   = q16 + NQ;
  unsigned short* apre16 = kv16 + NKV;
  unsigned short* wq_bf  = apre16 + NQ;
  unsigned short* wkv_bf = wq_bf + (size_t)512 * 512;
  unsigned short* w_bf   = wkv_bf + (size_t)512 * 1024;
  unsigned short* wp_bf  = w_bf + (size_t)512 * 512;

  // one-shot weight swizzle to bf16 fragment order (L2-resident afterwards)
  swizzle_weights<<<64, 256, 0, stream>>>(Wq, wq_bf, 512, 512);
  swizzle_weights<<<128, 256, 0, stream>>>(Wkv, wkv_bf, 512, 1024);
  swizzle_weights<<<64, 256, 0, stream>>>(W, w_bf, 512, 512);
  swizzle_weights<<<64, 256, 0, stream>>>(Wp, wp_bf, 512, 512);

  // q = noise @ Wq (bf16 out)
  gemm_af32_cbf16<<<dim3(8, 136), 256, 0, stream>>>(noise, wq_bf, q16, 512, 512, 512);
  // kv = obj @ Wkv (bf16 out)
  gemm_af32_cbf16<<<dim3(16, 512), 256, 0, stream>>>(obj, wkv_bf, kv16, 512, 512, 1024);
  // a_pre[b,n,h*64+m] = 0.125 * q.kT (bf16 out)
  attn_logits<<<1024, 256, 0, stream>>>(q16, kv16, apre16);
  // a = a_pre @ W (f32 out for BN stats)
  gemm_abf16_cf32<<<dim3(8, 136), 256, 0, stream>>>(apre16, w_bf, ws_a, 512, 512, 512);
  // BN batch stats -> per-channel scale/shift
  bn_stats<<<8, 256, 0, stream>>>(ws_a, gamma, beta, ws_scale, ws_shift);
  // BN apply + adjacency + softmax + attn@v + @Wp + bp
  fused_out<<<1024, 256, 0, stream>>>(ws_a, kv16, ws_scale, ws_shift, wp_bf, bp, out);

  (void)in_sizes; (void)n_in; (void)out_size;
}
